// MGU_20804821582085
// MI455X (gfx1250) — matrix operand order repaired and checked by reading, not compiled, not run on hardware
//
#include <hip/hip_runtime.h>
#include <hip/hip_bf16.h>

// ---------------------------------------------------------------------------
// MGU recurrence on MI455X (gfx1250), wave32 + V_WMMA_F32_16X16X32_BF16.
//
// T=512, B=64, I=H=512.  Per-step GEMM: [64 x 1024] = [64 x (512|512)] x W^T.
// 32 persistent workgroups, each owns 16 hidden columns j (gates j and H+j),
// keeps its 32 weight rows (K=1024, bf16) in 64 KB of LDS (XOR-swizzled).
// Per-step sync: all-to-all flag barrier (one L2 round trip).
// B-fragment pipeline is pinned with sched_group_barrier so ds_load stays
// >= 2 iterations ahead of its consuming v_wmma (no dscnt==0 stalls).
// ---------------------------------------------------------------------------

#define T_  512
#define B_  64
#define I_  512
#define H_  512
#define NWG 32          // == warpSize; each WG handles H_/NWG = 16 columns
#define NCOL 16
#define FLAG_STRIDE 16  // 64B spacing between per-WG flags

typedef __attribute__((ext_vector_type(16))) __bf16 v16bf;
typedef __attribute__((ext_vector_type(8)))  float  v8f;

__device__ __forceinline__ unsigned short f2bf(float f) {
    union { float f; unsigned u; } v; v.f = f;
    unsigned u = v.u;
    unsigned r = (u + 0x7FFFu + ((u >> 16) & 1u)) >> 16;   // round-nearest-even
    return (unsigned short)r;
}

__device__ __forceinline__ float fast_tanh(float x) {
#if __has_builtin(__builtin_amdgcn_tanhf)
    return __builtin_amdgcn_tanhf(x);       // gfx1250 v_tanh_f32
#else
    return tanhf(x);
#endif
}

__device__ __forceinline__ float fast_sigmoid(float x) {
    return 1.0f / (1.0f + __expf(-x));
}

__device__ __forceinline__ v8f wmma_bf16(v16bf a, v16bf b, v8f c) {
    // (neg_a, A, neg_b, B, c_mod, C, reuse_a, reuse_b)
    return __builtin_amdgcn_wmma_f32_16x16x32_bf16(
        false, a, false, b, (short)0, c, false, false);
}

// B fragment (32x16 bf16): lane reads 16 consecutive K halves = one 32B chunk
__device__ __forceinline__ v16bf ldsB(const unsigned short* wlds,
                                      int r, int ks, int kHi) {
    const int c = ks * 2 + kHi;                       // 32B chunk index
    const uint4* p = (const uint4*)&wlds[r * 1024 + ((c ^ r) << 4)];
    union { v16bf v; uint4 q[2]; } u;
    u.q[0] = p[0]; u.q[1] = p[1];
    return u.v;
}

// A fragment (16x32 bf16): lane reads halves [kA+off..+15]
__device__ __forceinline__ v16bf ldA(const unsigned short* base,
                                     int rowA, int kA, int offA) {
    const unsigned short* ap = base + (size_t)rowA * 512 + kA + offA;
    union { v16bf v; uint4 q[2]; } u;
    u.q[0] = *(const uint4*)(ap);
    u.q[1] = *(const uint4*)(ap + 8);
    return u.v;
}

// -------------------- kernel 1: fp32 -> bf16 streaming convert --------------
__global__ void cvt_f32_bf16(const float* __restrict__ src,
                             unsigned short* __restrict__ dst, int n4) {
    int i = blockIdx.x * blockDim.x + threadIdx.x;
    int stride = gridDim.x * blockDim.x;
    for (; i < n4; i += stride) {
        float4 v = ((const float4*)src)[i];
        ushort4 o;
        o.x = f2bf(v.x); o.y = f2bf(v.y); o.z = f2bf(v.z); o.w = f2bf(v.w);
        ((ushort4*)dst)[i] = o;
    }
}

// ---------- kernel 2: zero h0 (both parities) + barrier flags ---------------
__global__ void init_state(unsigned short* __restrict__ hbuf,
                           unsigned int* __restrict__ flags) {
    int i = blockIdx.x * blockDim.x + threadIdx.x;
    if (i < 2 * B_ * H_) hbuf[i] = 0;
    if (i < NWG * FLAG_STRIDE) flags[i] = 0u;
}

// -------------------- kernel 3: persistent recurrence -----------------------
__global__ __launch_bounds__(128, 1)
void mgu_persistent(const float* __restrict__ Wih,   // [2H, I]
                    const float* __restrict__ Whh,   // [2H, H]
                    const float* __restrict__ bih,   // [2H]
                    const float* __restrict__ bhh,   // [2H]
                    const unsigned short* __restrict__ xbf,  // [T,B,I] bf16
                    unsigned short* __restrict__ hbuf,       // [2,B,H] bf16
                    float* __restrict__ out,                 // [T,B,H] f32
                    unsigned int* __restrict__ flags) {
    // 32 weight rows x 1024 K halves, 32B-chunk XOR swizzle -> exactly 64 KB.
    __shared__ unsigned short wlds[32 * 1024];

    const int tid  = threadIdx.x;
    const int lane = tid & 31;
    const int wv   = tid >> 5;          // 0..3, one M-tile per wave
    const int wg   = blockIdx.x;
    const int j0   = wg * NCOL;

    // ---- stage this WG's weight rows into LDS (once), fp32 -> bf16 ----
    // row r: 0..15 -> f-gate j0+r ; 16..31 -> n-gate H+j0+(r-16)
    // K layout per row: [0,512) = W_ih row, [512,1024) = W_hh row.
    for (int ch = tid; ch < 32 * 64; ch += 128) {
        int r = ch >> 6, c = ch & 63;                       // chunk = 16 halves
        int g = (r < 16) ? (j0 + r) : (H_ + j0 + (r - 16));
        const float* src = (c < 32) ? (Wih + (size_t)g * I_ + c * 16)
                                    : (Whh + (size_t)g * H_ + (c - 32) * 16);
        int dst = r * 1024 + ((c ^ r) << 4);                // XOR swizzle
#pragma unroll
        for (int e = 0; e < 16; ++e) wlds[dst + e] = f2bf(src[e]);
    }
    __syncthreads();

    // ---- per-lane constants ----
    const int n_  = lane & 15;          // C-layout: lane -> column
    const int jf  = j0 + n_;            // forget-gate column (global)
    const int jn  = H_ + jf;            // new-gate column (global)
    const float bfs = bih[jf] + bhh[jf];    // merged forget bias
    const float bin = bih[jn];              // i_n bias
    const float bhn = bhh[jn];              // h_n bias

    const int rowA = wv * 16 + n_;              // A-layout: lane -> M row
    const int offA = (lane & 16) ? 16 : 0;      // A k-offset per half-wave
    const int kHi  = (lane >> 4) & 1;           // B k-chunk per half-wave
    const int rF   = n_;                        // B row (f-gate) in LDS
    const int rN   = 16 + n_;                   // B row (n-gate) in LDS

    for (int t = 0; t < T_; ++t) {
        const unsigned short* xA = xbf + (size_t)t * B_ * I_;
        const unsigned short* hA = hbuf + (size_t)(t & 1) * B_ * H_;
        unsigned short* hN = hbuf + (size_t)((t + 1) & 1) * B_ * H_;

        // pull freshly-written h row (1 KB/lane) toward this WGP right away;
        // it is consumed ~an x-phase later, hiding the L2 hop.
        __builtin_prefetch(hA + (size_t)rowA * H_,       0, 3);
        __builtin_prefetch(hA + (size_t)rowA * H_ + 128, 0, 3);
        __builtin_prefetch(hA + (size_t)rowA * H_ + 256, 0, 3);
        __builtin_prefetch(hA + (size_t)rowA * H_ + 384, 0, 3);
        if (t + 1 < T_) {
            const unsigned short* xn = xA + (size_t)B_ * I_;
            __builtin_prefetch(xn + (size_t)rowA * I_,       0, 3);
            __builtin_prefetch(xn + (size_t)rowA * I_ + 256, 0, 3);
        }

        v8f accf  = {};     // i_f + h_f   (K = 1024 merged)
        v8f accin = {};     // i_n         (K = 512, x part)
        v8f acchn = {};     // h_n         (K = 512, h part)

        // ---- merged K loop, distance-2 pipelined B fragments ---------------
        // ks 0..15: A = x_t (W_ih half)   ks 16..31: A = h (W_hh half)
        v16bf bF[3], bN[3];
        bF[0] = ldsB(wlds, rF, 0, kHi);  bN[0] = ldsB(wlds, rN, 0, kHi);
        bF[1] = ldsB(wlds, rF, 1, kHi);  bN[1] = ldsB(wlds, rN, 1, kHi);
#pragma unroll
        for (int ks = 0; ks < 32; ++ks) {
            const int cur = ks % 3;
            const int nxt = (ks + 2) % 3;
            if (ks < 30) {                      // fetch for ks+2 (2 ahead)
                bF[nxt] = ldsB(wlds, rF, ks + 2, kHi);
                bN[nxt] = ldsB(wlds, rN, ks + 2, kHi);
            }
            v16bf a = (ks < 16) ? ldA(xA, rowA, ks * 32, offA)
                                : ldA(hA, rowA, (ks - 16) * 32, offA);
            accf = wmma_bf16(a, bF[cur], accf);
            if (ks < 16) accin = wmma_bf16(a, bN[cur], accin);
            else         acchn = wmma_bf16(a, bN[cur], acchn);
        }

#if __has_builtin(__builtin_amdgcn_sched_group_barrier)
        // Pin the pipeline: [8 DS-read] then 30x{ [2 WMMA] [4 DS-read] }
        // then [4 WMMA].  DS-read = 0x100, MFMA/WMMA = 0x8.
        // Per step: 128 ds_load_b128 (64 B frags) and 64 v_wmma -- exact fit.
        __builtin_amdgcn_sched_group_barrier(0x100, 8, 0);
#pragma unroll
        for (int g = 0; g < 30; ++g) {
            __builtin_amdgcn_sched_group_barrier(0x008, 2, 0);
            __builtin_amdgcn_sched_group_barrier(0x100, 4, 0);
        }
        __builtin_amdgcn_sched_group_barrier(0x008, 4, 0);
#endif

        // ---- activations + stores (all in registers; bias is lane-scalar) --
#pragma unroll
        for (int i = 0; i < 8; ++i) {
            int b = wv * 16 + i + ((lane & 16) ? 8 : 0);    // C-layout M row
            float f   = fast_sigmoid(accf[i] + bfs);
            float val = fast_tanh((accin[i] + bin) + f * (acchn[i] + bhn));
            out[(size_t)t * B_ * H_ + (size_t)b * H_ + jf] = val;
            hN[(size_t)b * H_ + jf] = f2bf(val);
        }

        // ---- all-to-all flag barrier: 1 release store + parallel poll ------
        if (t + 1 < T_) {
            __threadfence();                    // publish h stores (agent)
            __syncthreads();                    // whole WG done storing
            if (wv == 0) {                      // wave 0: signal + poll
                const unsigned target = (unsigned)(t + 1);
                if (lane == 0)
                    __hip_atomic_store(&flags[wg * FLAG_STRIDE], target,
                                       __ATOMIC_RELEASE,
                                       __HIP_MEMORY_SCOPE_AGENT);
                for (;;) {                      // lane l watches WG l's flag
                    unsigned v = __hip_atomic_load(&flags[lane * FLAG_STRIDE],
                                                   __ATOMIC_ACQUIRE,
                                                   __HIP_MEMORY_SCOPE_AGENT);
                    if (__all(v >= target)) break;
                    __builtin_amdgcn_s_sleep(1);
                }
            }
            __syncthreads();                    // broadcast barrier-done
            __threadfence();                    // acquire side for h loads
        }
    }
}

// ---------------------------------------------------------------------------
extern "C" void kernel_launch(void* const* d_in, const int* in_sizes, int n_in,
                              void* d_out, int out_size, void* d_ws, size_t ws_size,
                              hipStream_t stream) {
    const float* x   = (const float*)d_in[0];   // (T,B,I)
    const float* wih = (const float*)d_in[1];   // (2H,I)
    const float* whh = (const float*)d_in[2];   // (2H,H)
    const float* bih = (const float*)d_in[3];   // (2H)
    const float* bhh = (const float*)d_in[4];   // (2H)
    float* out = (float*)d_out;                 // (T,B,H)

    // ws layout: [ x_bf16 : T*B*I halves | h_bf16 : 2*B*H halves | flags ]
    unsigned short* xbf   = (unsigned short*)d_ws;
    unsigned short* hbuf  = xbf + (size_t)T_ * B_ * I_;
    unsigned int*   flags = (unsigned int*)(hbuf + (size_t)2 * B_ * H_);

    // 1) convert input activations to bf16 (one streaming pass, ~100 MB total)
    const int n4 = (T_ * B_ * I_) / 4;
    cvt_f32_bf16<<<2048, 256, 0, stream>>>(x, xbf, n4);

    // 2) zero h0 (both parities) + barrier flags, every call (determinism)
    init_state<<<(2 * B_ * H_ + 255) / 256, 256, 0, stream>>>(hbuf, flags);

    // 3) persistent fused recurrence: 32 WGs x 128 threads, 64 KB LDS each
    mgu_persistent<<<NWG, 128, 0, stream>>>(wih, whh, bih, bhh,
                                            xbf, hbuf, out, flags);
}
